// LinearAttention_80719615361342
// MI455X (gfx1250) — compile-verified, matrix operand order checked by reading, not run on hardware
//
#include <hip/hip_runtime.h>

// ---------------------------------------------------------------- types
typedef __bf16 bf16;
typedef __attribute__((ext_vector_type(16))) __bf16 bf16x16;
typedef __attribute__((ext_vector_type(8)))  float  f32x8;

// ---------------------------------------------------------------- sizes
constexpr int BB   = 4;            // batch
constexpr int SS   = 4096;         // sequence
constexpr int DD   = 1024;         // d_model
constexpr int HH   = 16;           // heads
constexpr int HD   = 64;           // head_dim
constexpr int FF   = 256;          // feature dim
constexpr int MTOT = BB * SS;      // 16384 rows
constexpr float EPSV = 1e-8f;

// ------------------------------------------------ CDNA5 async global->LDS DMA
// global_load_async_to_lds_b128: VDST = LDS byte offset (VGPR), VADDR = 64-bit
// global address.  Tracked with ASYNCcnt (CDNA5 ISA).
__device__ __forceinline__ void async_copy16(unsigned lds_off, const void* gptr) {
  asm volatile("global_load_async_to_lds_b128 %0, %1, off"
               :: "v"(lds_off), "v"(gptr) : "memory");
}
__device__ __forceinline__ void wait_async_all() {
  asm volatile("s_wait_asynccnt 0x0" ::: "memory");
}

// ================================================================ fp32 -> bf16
__global__ void cvt_f32_bf16(const float* __restrict__ src,
                             bf16* __restrict__ dst, int n4) {
  int i = blockIdx.x * blockDim.x + threadIdx.x;
  if (i >= n4) return;
  float4 v = reinterpret_cast<const float4*>(src)[i];
  union { bf16 h[4]; uint2 u; } p;
  p.h[0] = (bf16)v.x; p.h[1] = (bf16)v.y; p.h[2] = (bf16)v.z; p.h[3] = (bf16)v.w;
  reinterpret_cast<uint2*>(dst)[i] = p.u;
}

// ================================================================ fp32 -> bf16 transpose
// dst[N][K] = (bf16) src[K][N].   grid (K/32, N/32), block (32, 8).
__global__ void cvt_transpose(const float* __restrict__ src, bf16* __restrict__ dst,
                              int K, int N) {
  __shared__ float tile[32][33];
  int k0 = blockIdx.x * 32, n0 = blockIdx.y * 32;
  int tx = threadIdx.x, ty = threadIdx.y;
#pragma unroll
  for (int i = 0; i < 4; ++i)
    tile[ty + 8 * i][tx] = src[(k0 + ty + 8 * i) * N + n0 + tx];
  __syncthreads();
#pragma unroll
  for (int i = 0; i < 4; ++i)
    dst[(n0 + ty + 8 * i) * K + k0 + tx] = (bf16)tile[tx][ty + 8 * i];
}

// ================================================================ WMMA GEMM
// C = act( A[M,K] @ Bt[N,K]^T + bias[N] ),  A and Bt both row-major along K so
// every LDS stage is an async B128 row copy.  TRANS_C writes C transposed
// (C[n*ldc + m]) to produce vT / kfT / kvT for downstream consumers.
// Batched over gridDim.z: z -> (b = z/Hdiv, h = z%Hdiv) with int strides.
// Tile 128x64x32, 256 threads = 8 waves, wave = 32x32 (2x2 WMMA frags).
// Double-buffered LDS, async copy of tile kt+1 overlaps WMMAs of tile kt.
template <bool TRANS_C, bool RELU, bool OUT_F32>
__global__ __launch_bounds__(256)
void gemm_wmma(const bf16* __restrict__ A,  int lda,  int sAb, int sAh,
               const bf16* __restrict__ Bt, int ldbt, int sBb, int sBh,
               const float* __restrict__ bias,
               void* __restrict__ Cp, int ldc, int sCb, int sCh,
               int K, int Hdiv) {
  constexpr int BM = 128, BN = 64, BK = 32, LP = 40;  // LP: bank-safe LDS row
  __shared__ __align__(16) bf16 As[2][BM][LP];
  __shared__ __align__(16) bf16 Bs[2][BN][LP];

  const int z  = blockIdx.z;
  const int bb = z / Hdiv, hh = z % Hdiv;
  A  += bb * sAb + hh * sAh;
  Bt += bb * sBb + hh * sBh;
  const int cOff = bb * sCb + hh * sCh;

  const int m0   = blockIdx.x * BM;
  const int n0   = blockIdx.y * BN;
  const int tid  = threadIdx.x;
  const int wave = tid >> 5, lane = tid & 31;
  const int hlf  = lane >> 4, lr = lane & 15;
  const int wm   = (wave >> 1) * 32;      // 8 waves -> 4x2 macro-tiles
  const int wn   = (wave & 1)  * 32;

  // per-thread staging assignment (A: 512 16B units / 256 thr; B: 256 units)
  const int am  = tid >> 1, aku = (tid & 1) * 16;   // A row, 2 units (32B)
  const int bn  = tid >> 2, bku = (tid & 3) * 8;    // B row, 1 unit (16B)
  const bf16* aNxt = A  + (m0 + am) * lda  + aku;
  const bf16* bNxt = Bt + (n0 + bn) * ldbt + bku;
  const unsigned aDst[2] = { (unsigned)(unsigned long long)&As[0][am][aku],
                             (unsigned)(unsigned long long)&As[1][am][aku] };
  const unsigned bDst[2] = { (unsigned)(unsigned long long)&Bs[0][bn][bku],
                             (unsigned)(unsigned long long)&Bs[1][bn][bku] };

  f32x8 acc[2][2] = {};
  const int kTiles = K / BK;

  // prologue: DMA tile 0 into buffer 0
  async_copy16(aDst[0],      aNxt);
  async_copy16(aDst[0] + 16, aNxt + 8);
  async_copy16(bDst[0],      bNxt);
  aNxt += BK; bNxt += BK;

  for (int kt = 0; kt < kTiles; ++kt) {
    wait_async_all();        // this wave's DMA for tile kt done
    __syncthreads();         // everyone's DMA visible; prev compute finished
    if (kt + 1 < kTiles) {   // overlap next tile's DMA with this tile's WMMAs
      const int nb = (kt + 1) & 1;
      async_copy16(aDst[nb],      aNxt);
      async_copy16(aDst[nb] + 16, aNxt + 8);
      async_copy16(bDst[nb],      bNxt);
      aNxt += BK; bNxt += BK;
    }
    const int cur = kt & 1;

    // fragments (ISA 16-bit A/B 16x32 layout):
    // lanes 0-15: K 0..7,16..23 ; lanes 16-31: K 8..15,24..31
    bf16x16 af[2], bfr[2];
    const int kb = hlf * 8;
#pragma unroll
    for (int t = 0; t < 2; ++t) {
      const bf16* ar = &As[cur][wm + t * 16 + lr][0];
      const bf16* br = &Bs[cur][wn + t * 16 + lr][0];
#pragma unroll
      for (int i = 0; i < 8; ++i) {
        af[t][i]      = ar[kb + i];
        af[t][i + 8]  = ar[16 + kb + i];
        bfr[t][i]     = br[kb + i];
        bfr[t][i + 8] = br[16 + kb + i];
      }
    }
#pragma unroll
    for (int tm = 0; tm < 2; ++tm)
#pragma unroll
      for (int tn = 0; tn < 2; ++tn)
        acc[tm][tn] = __builtin_amdgcn_wmma_f32_16x16x32_bf16(
            false, af[tm], false, bfr[tn], (short)0, acc[tm][tn], false, false);
  }

  // epilogue: C frag VGPR r -> M = r (lanes 0-15) / 8+r (lanes 16-31), N = lr
#pragma unroll
  for (int tm = 0; tm < 2; ++tm)
#pragma unroll
    for (int tn = 0; tn < 2; ++tn)
#pragma unroll
      for (int r = 0; r < 8; ++r) {
        int mg = m0 + wm + tm * 16 + hlf * 8 + r;
        int ng = n0 + wn + tn * 16 + lr;
        float v = acc[tm][tn][r];
        if (bias) v += bias[ng];
        if (RELU) v = fmaxf(v, 0.f);
        int addr = TRANS_C ? (cOff + ng * ldc + mg) : (cOff + mg * ldc + ng);
        if (OUT_F32) reinterpret_cast<float*>(Cp)[addr] = v;
        else         reinterpret_cast<bf16*>(Cp)[addr] = (bf16)v;
      }
}

// ============================================== k_sum[b,h,f] = sum_s kfT (contiguous!)
__global__ void ksum_kernel(const bf16* __restrict__ kfT, float* __restrict__ ks) {
  int idx = blockIdx.x * blockDim.x + threadIdx.x;   // B*H*F = 16384
  if (idx >= BB * HH * FF) return;
  int f = idx % FF, h = (idx / FF) % HH, b = idx / (FF * HH);
  const bf16* p = kfT + h * (FF * MTOT) + f * MTOT + b * SS;
  float s = 0.f;
  for (int t = 0; t < SS / 8; ++t) {
    union { uint4 u; bf16 hv[8]; } w;
    w.u = reinterpret_cast<const uint4*>(p)[t];
#pragma unroll
    for (int j = 0; j < 8; ++j) s += (float)w.hv[j];
  }
  ks[idx] = s;
}

// ============================================== qks[b,s,h] = qf . k_sum
__global__ void qks_kernel(const bf16* __restrict__ qf, const float* __restrict__ ks,
                           float* __restrict__ qks) {
  int idx = blockIdx.x * blockDim.x + threadIdx.x;   // B*S*H = 262144
  if (idx >= MTOT * HH) return;
  int h  = idx % HH;
  int bs = idx / HH;
  int b  = bs / SS;
  const bf16*  q  = qf + bs * (HH * FF) + h * FF;
  const float* kp = ks + (b * HH + h) * FF;
  float a = 0.f;
  for (int t = 0; t < FF / 8; ++t) {
    union { uint4 u; bf16 hv[8]; } w;
    w.u = reinterpret_cast<const uint4*>(q)[t];
#pragma unroll
    for (int j = 0; j < 8; ++j) a += (float)w.hv[j] * kp[t * 8 + j];
  }
  qks[idx] = a;
}

// ============================================== normalize + cast
__global__ void norm_kernel(const float* __restrict__ out_un,
                            const float* __restrict__ qks,
                            bf16* __restrict__ out_pre) {
  int idx = blockIdx.x * blockDim.x + threadIdx.x;   // MTOT*DD
  if (idx >= MTOT * DD) return;
  int row = idx / DD, col = idx % DD, h = col / HD;
  float den = qks[row * HH + h] + EPSV;
  out_pre[idx] = (bf16)(out_un[idx] / den);
}

// ================================================================ launch
extern "C" void kernel_launch(void* const* d_in, const int* in_sizes, int n_in,
                              void* d_out, int out_size, void* d_ws, size_t ws_size,
                              hipStream_t stream) {
  const float* query = (const float*)d_in[0];
  const float* key_  = (const float*)d_in[1];
  const float* value = (const float*)d_in[2];
  const float* Wq = (const float*)d_in[3];  const float* bq = (const float*)d_in[4];
  const float* Wk = (const float*)d_in[5];  const float* bk = (const float*)d_in[6];
  const float* Wv = (const float*)d_in[7];  const float* bv = (const float*)d_in[8];
  const float* Wf = (const float*)d_in[9];  const float* bf_ = (const float*)d_in[10];
  const float* Wo = (const float*)d_in[11]; const float* bo = (const float*)d_in[12];

  size_t off = 0;
  auto alloc = [&](size_t bytes) -> char* {
    char* p = (char*)d_ws + off;
    off += (bytes + 255) & ~(size_t)255;
    return p;
  };
  bf16* qin   = (bf16*)alloc((size_t)MTOT * DD * 2);
  bf16* kin   = (bf16*)alloc((size_t)MTOT * DD * 2);
  bf16* vin   = (bf16*)alloc((size_t)MTOT * DD * 2);
  bf16* WqT   = (bf16*)alloc((size_t)DD * DD * 2);
  bf16* WkT   = (bf16*)alloc((size_t)DD * DD * 2);
  bf16* WvT   = (bf16*)alloc((size_t)DD * DD * 2);
  bf16* WoT   = (bf16*)alloc((size_t)DD * DD * 2);
  bf16* WfT   = (bf16*)alloc((size_t)HD * FF * 2);
  bf16* qproj = (bf16*)alloc((size_t)MTOT * DD * 2);
  bf16* kproj = (bf16*)alloc((size_t)MTOT * DD * 2);
  bf16* vT    = (bf16*)alloc((size_t)MTOT * DD * 2);        // [D][B*S]
  bf16* qf    = (bf16*)alloc((size_t)MTOT * HH * FF * 2);   // [(b,s)][h*F+f]
  bf16* kfT   = (bf16*)alloc((size_t)MTOT * HH * FF * 2);   // [h][f][(b,s)]
  bf16* kvT   = (bf16*)alloc((size_t)BB * HH * HD * FF * 2);// [b,h][d][f]
  float* ksum = (float*)alloc((size_t)BB * HH * FF * 4);
  float* qks  = (float*)alloc((size_t)MTOT * HH * 4);
  float* outu = (float*)alloc((size_t)MTOT * DD * 4);
  bf16* outp  = (bf16*)alloc((size_t)MTOT * DD * 2);

  auto cvt = [&](const float* s, bf16* d, int n) {
    int n4 = n / 4;
    cvt_f32_bf16<<<dim3((n4 + 255) / 256), dim3(256), 0, stream>>>(s, d, n4);
  };
  cvt(query, qin, MTOT * DD);
  cvt(key_,  kin, MTOT * DD);
  cvt(value, vin, MTOT * DD);
  cvt_transpose<<<dim3(DD / 32, DD / 32), dim3(32, 8), 0, stream>>>(Wq, WqT, DD, DD);
  cvt_transpose<<<dim3(DD / 32, DD / 32), dim3(32, 8), 0, stream>>>(Wk, WkT, DD, DD);
  cvt_transpose<<<dim3(DD / 32, DD / 32), dim3(32, 8), 0, stream>>>(Wv, WvT, DD, DD);
  cvt_transpose<<<dim3(DD / 32, DD / 32), dim3(32, 8), 0, stream>>>(Wo, WoT, DD, DD);
  cvt_transpose<<<dim3(HD / 32, FF / 32), dim3(32, 8), 0, stream>>>(Wf, WfT, HD, FF);

  const dim3 blk(256);
  // --- q/k projections: C normal;  v projection: C transposed -> vT[D][MTOT]
  {
    dim3 g(MTOT / 128, DD / 64, 1);
    gemm_wmma<false, false, false><<<g, blk, 0, stream>>>(
        qin, DD, 0, 0, WqT, DD, 0, 0, bq, qproj, DD, 0, 0, DD, 1);
    gemm_wmma<false, false, false><<<g, blk, 0, stream>>>(
        kin, DD, 0, 0, WkT, DD, 0, 0, bk, kproj, DD, 0, 0, DD, 1);
    gemm_wmma<true, false, false><<<g, blk, 0, stream>>>(
        vin, DD, 0, 0, WvT, DD, 0, 0, bv, vT, MTOT, 0, 0, DD, 1);
  }
  // --- feature maps per head: qf normal, kf transposed -> kfT[h][f][(b,s)]
  {
    dim3 g(MTOT / 128, FF / 64, HH);
    gemm_wmma<false, true, false><<<g, blk, 0, stream>>>(
        qproj, DD, 0, HD, WfT, HD, 0, 0, bf_, qf, HH * FF, 0, FF, HD, HH);
    gemm_wmma<true, true, false><<<g, blk, 0, stream>>>(
        kproj, DD, 0, HD, WfT, HD, 0, 0, bf_, kfT, MTOT, 0, FF * MTOT, HD, HH);
  }
  // --- kv[b,h]: [F,hd] = kfT rows @ vT rows^T over K=S; write kvT[b,h][d][f]
  {
    dim3 g(FF / 128, HD / 64, BB * HH);
    gemm_wmma<true, false, false><<<g, blk, 0, stream>>>(
        kfT, MTOT, SS, FF * MTOT, vT, MTOT, SS, HD * MTOT, nullptr,
        kvT, FF, HH * HD * FF, HD * FF, SS, HH);
  }
  ksum_kernel<<<dim3((BB * HH * FF + 255) / 256), dim3(256), 0, stream>>>(kfT, ksum);
  // --- out_un[b,:,h,:]: [S,hd] = qf @ kvT^T over K=F
  {
    dim3 g(SS / 128, HD / 64, BB * HH);
    gemm_wmma<false, false, true><<<g, blk, 0, stream>>>(
        qf, HH * FF, SS * HH * FF, FF, kvT, FF, HH * HD * FF, HD * FF,
        nullptr, outu, DD, SS * DD, HD, FF, HH);
  }
  qks_kernel<<<dim3((MTOT * HH + 255) / 256), dim3(256), 0, stream>>>(qf, ksum, qks);
  norm_kernel<<<dim3((MTOT * DD + 255) / 256), dim3(256), 0, stream>>>(outu, qks, outp);
  // --- output projection: d_out (f32) = out_pre @ WoT^T + bo
  {
    dim3 g(MTOT / 128, DD / 64, 1);
    gemm_wmma<false, false, true><<<g, blk, 0, stream>>>(
        outp, DD, 0, 0, WoT, DD, 0, 0, bo, (float*)d_out, DD, 0, 0, DD, 1);
  }
}
